// CTLSTM_18296560681638
// MI455X (gfx1250) — matrix operand order repaired and checked by reading, not compiled, not run on hardware
//
#include <hip/hip_runtime.h>

// ---- problem constants --------------------------------------------------
#define Hsz   512
#define Bsz   32
#define Tlen  1024
#define MTOT  (Bsz * Tlen)          // 32768 rows (b*T + t)
#define GOUT  (Tlen * Bsz * Hsz)    // 16777216 elements per output slice
#define NW    (5 * Hsz * Hsz)       // packed bf16 weight element count
#define NTAB  ((1024 + 1) * Hsz)    // bf16 embedding table element count

// ---- tiling -------------------------------------------------------------
#define MT    128                   // rows per workgroup
#define HT    128                   // h-columns per workgroup
#define KC    64                    // K chunk staged in LDS
#define ESTR  72                    // padded LDS k-stride (ushorts): 64 data + 8 pad
#define EMBSZ (MT * ESTR)           // 9216  ushorts per emb buffer
#define WSZ   (5 * HT * ESTR)       // 46080 ushorts per weight buffer
#define BUFSZ (EMBSZ + WSZ)         // 55296 ushorts per double-buffer stage

typedef __attribute__((ext_vector_type(16))) __bf16 v16bf;
typedef __attribute__((ext_vector_type(8)))  __bf16 v8bf;
typedef __attribute__((ext_vector_type(8)))  float  v8f;
typedef __attribute__((ext_vector_type(4)))  unsigned int v4u;
typedef __attribute__((ext_vector_type(8)))  int v8i;
typedef __attribute__((ext_vector_type(4)))  int v4i;

#if defined(__has_builtin)
#if __has_builtin(__builtin_amdgcn_tensor_load_to_lds)
#define HAS_TDM 1
#endif
#endif
#if __has_include(<hip/amd_detail/amd_gfx1250_TDM.h>)
#define TDM_6ARG 1
#endif

static __device__ __forceinline__ unsigned short f2bf(float x) {
    unsigned int u = __float_as_uint(x);
    u += 0x7FFFu + ((u >> 16) & 1u);      // round-to-nearest-even
    return (unsigned short)(u >> 16);
}

// Convert the 5 *used* gate chunks of W_rec[:, :H] (gi, gz, go, gib, gd) to bf16.
// Layout: Wb[g][n][k], g-major, n,k in [0,512).
__global__ void cvt_w(const float* __restrict__ W, unsigned short* __restrict__ dst) {
    int i = blockIdx.x * 256 + threadIdx.x;
    if (i >= NW) return;
    int g = i >> 18;
    int r = (i >> 9) & 511;
    int k = i & 511;
    const int goff[5] = {0, 2 * Hsz, 3 * Hsz, 4 * Hsz, 6 * Hsz};
    dst[i] = f2bf(W[(size_t)(goff[g] + r) * (2 * Hsz) + k]);
}

__global__ void cvt_t(const float* __restrict__ src, unsigned short* __restrict__ dst, int n) {
    int i = blockIdx.x * 256 + threadIdx.x;
    if (i < n) dst[i] = f2bf(src[i]);
}

static __device__ __forceinline__ v16bf cat16(v8bf lo, v8bf hi) {
    return __builtin_shufflevector(lo, hi, 0, 1, 2, 3, 4, 5, 6, 7,
                                           8, 9, 10, 11, 12, 13, 14, 15);
}

static __device__ __forceinline__ float fsig(float x) {
    return __builtin_amdgcn_rcpf(1.f + __expf(-x));
}
static __device__ __forceinline__ float ftanh(float x) {
    // 1 - 2/(e^{2x}+1): saturates correctly to +-1 at large |x|
    return 1.f - 2.f * __builtin_amdgcn_rcpf(__expf(2.f * x) + 1.f);
}

#ifdef HAS_TDM
// TDM: DMA a 3D bf16 tile (x=KC cols, y=HT rows stride Hsz, z=5 gates stride Hsz*Hsz)
// from global Wb into LDS, inserting 16B of padding after every 128B row so the
// LDS image lands directly in the conflict-free [5][HT][ESTR] layout.
static __device__ __forceinline__ void tdm_load_weights(unsigned int ldsByteOff,
                                                        const unsigned short* gsrc) {
    unsigned long long ga = (unsigned long long)(size_t)gsrc;
    v4u g0;
    g0[0] = 1u;                                      // count=1, no gather
    g0[1] = ldsByteOff;                              // lds_addr
    g0[2] = (unsigned int)ga;                        // global_addr[31:0]
    g0[3] = (unsigned int)((ga >> 32) & 0x01FFFFFFull) | (2u << 30); // addr[56:32] | type=2
    v8i g1;
    g1[0] = (1 << 16)        // data_size = 2B
          | (1 << 20)        // pad_enable
          | (4 << 22)        // pad_interval: 32 DWORDs (= 128B row)
          | (3 << 25);       // pad_amount: 4 DWORDs (= 16B pad)
    g1[1] = (int)(512u << 16);          // tensor_dim0 = 512 (low 16 @ bits 63:48)
    g1[2] = (int)(512u << 16);          // dim0 hi=0; tensor_dim1 = 512 (low 16 @ 95:80)
    g1[3] = (int)((unsigned)KC << 16);  // dim1 hi=0; tile_dim0 = 64
    g1[4] = (int)((unsigned)HT | (5u << 16)); // tile_dim1 = 128, tile_dim2 = 5
    g1[5] = Hsz;                        // tensor_dim0_stride = 512 (low 32)
    g1[6] = 0;                          // dim0_stride hi; dim1_stride low16 = 0
    g1[7] = (Hsz * Hsz) >> 16;          // tensor_dim1_stride = 262144 (>>16 = 4)
    v4i g2;
    g2[0] = 5;                          // tensor_dim2 = 5
    g2[1] = 0; g2[2] = 0; g2[3] = 0;    // no dim3, tile_dim3 = 0
    v4i g3 = {0, 0, 0, 0};
#ifdef TDM_6ARG
    v8i g4 = {0, 0, 0, 0, 0, 0, 0, 0};
    __builtin_amdgcn_tensor_load_to_lds(g0, g1, g2, g3, g4, 0);
#else
    __builtin_amdgcn_tensor_load_to_lds(g0, g1, g2, g3, 0);
#endif
}
#endif

__global__ void __launch_bounds__(512, 1)
ctlstm_fused(const int* __restrict__ ev, const float* __restrict__ dur,
             const float* __restrict__ brec,
             const unsigned short* __restrict__ tabB,
             const unsigned short* __restrict__ Wb,
             float* __restrict__ out) {
    extern __shared__ __align__(16) unsigned short smem[]; // 2 x BUFSZ ushorts

    const int tid    = threadIdx.x;
    const int lane   = tid & 31;
    const int wid    = tid >> 5;       // 0..15
    const int mw     = wid & 3;        // wave's 32-row group
    const int hw     = wid >> 2;       // wave's 32-col group
    const int lane15 = lane & 15;
    const int khalf  = lane >> 4;      // 0 / 1
    const int m0     = blockIdx.x * MT;
    const int h0     = blockIdx.y * HT;
    const unsigned int smemBase = (unsigned int)(size_t)(void*)smem;

    v8f acc[2][2][5];
#pragma unroll
    for (int mt = 0; mt < 2; ++mt)
#pragma unroll
        for (int ht = 0; ht < 2; ++ht)
#pragma unroll
            for (int g = 0; g < 5; ++g)
                acc[mt][ht][g] = v8f{0.f, 0.f, 0.f, 0.f, 0.f, 0.f, 0.f, 0.f};

    // ---- embedding gather setup: 4 threads x 32B per row --------------------
    const int row = tid >> 2, q = tid & 3;
    const int idx = ev[m0 + row];
    const unsigned short* srcRow = tabB + (size_t)idx * Hsz + q * 16;

    // ---- prologue: prefetch chunk 0 -----------------------------------------
    uint4 r0, r1;
    {
        const uint4* s = (const uint4*)srcRow;
        r0 = s[0]; r1 = s[1];
    }
#ifdef HAS_TDM
    if (wid == 0)
        tdm_load_weights(smemBase + (unsigned)EMBSZ * 2u, Wb + (size_t)h0 * Hsz);
#endif

#pragma unroll 1
    for (int kc = 0; kc < Hsz; kc += KC) {
        const int cur = (kc >> 6) & 1;
        unsigned short* embB = smem + cur * BUFSZ;
        unsigned short* WbB  = embB + EMBSZ;

        // commit prefetched emb rows into current buffer
        {
            uint4* d = (uint4*)(embB + row * ESTR + q * 16);
            d[0] = r0; d[1] = r1;
        }
#ifdef HAS_TDM
        if (wid == 0) __builtin_amdgcn_s_wait_tensorcnt(0);
#else
        // fallback: cooperative weight staging (2560 units of 8 ushorts)
#pragma unroll
        for (int it = 0; it < 5; ++it) {
            int u = tid + it * 512;
            int g = u >> 9;
            int n = (u >> 2) & 127;
            int qq = u & 3;
            const uint4* s = (const uint4*)(Wb + ((size_t)g * Hsz + h0 + n) * Hsz + kc + qq * 16);
            *(uint4*)(WbB + (g * HT + n) * ESTR + qq * 16) = *s;
        }
#endif
        __syncthreads();

        // kick off next chunk's transfers while we compute this one
        if (kc + KC < Hsz) {
            const uint4* s = (const uint4*)(srcRow + kc + KC);
            r0 = s[0]; r1 = s[1];
#ifdef HAS_TDM
            if (wid == 0)
                tdm_load_weights(smemBase + ((unsigned)((1 - cur) * BUFSZ + EMBSZ)) * 2u,
                                 Wb + (size_t)h0 * Hsz + (kc + KC));
#endif
        }

        // ---- 40 WMMAs on the current chunk, B-fragment software pipeline ----
        // Load all 4 A fragments (2 k0 x 2 mt).
        // A (16x32 bf16): lanes 0-15 hold K 0-15, lanes 16-31 K 16-31.
        v16bf A[2][2];
#pragma unroll
        for (int k0i = 0; k0i < 2; ++k0i)
#pragma unroll
            for (int mt = 0; mt < 2; ++mt) {
                const unsigned short* p =
                    embB + (mw * 32 + mt * 16 + lane15) * ESTR + k0i * 32 + khalf * 16;
                v8bf lo = *(const v8bf*)p;
                v8bf hi = *(const v8bf*)(p + 8);
                A[k0i][mt] = cat16(lo, hi);
            }

        // B fragment loader: position p in [0,20) -> (k0i, ht, g).
        // B (32x16 bf16): lanes 0-15 hold K 0-15 of column n, lanes 16-31 K 16-31.
        const unsigned short* Wbase =
            WbB + (hw * 32 + lane15) * ESTR + khalf * 16;
        auto loadB = [&](int p) -> v16bf {
            const int k0i = p / 10, rem = p % 10, ht = rem / 5, g = rem % 5;
            const unsigned short* ptr =
                Wbase + (g * HT + ht * 16) * ESTR + k0i * 32;
            v8bf lo = *(const v8bf*)ptr;
            v8bf hi = *(const v8bf*)(ptr + 8);
            return cat16(lo, hi);
        };

        v16bf Bc = loadB(0);
#pragma unroll
        for (int p = 0; p < 20; ++p) {
            const int k0i = p / 10, rem = p % 10, ht = rem / 5, g = rem % 5;
            v16bf Bn = Bc;
            if (p < 19) Bn = loadB(p + 1);   // issue next LDS fetch before WMMAs
            acc[0][ht][g] = __builtin_amdgcn_wmma_f32_16x16x32_bf16(
                false, A[k0i][0], false, Bc, (short)0, acc[0][ht][g], false, false);
            acc[1][ht][g] = __builtin_amdgcn_wmma_f32_16x16x32_bf16(
                false, A[k0i][1], false, Bc, (short)0, acc[1][ht][g], false, false);
            Bc = Bn;
        }
        __syncthreads();
    }

    // ---- fused epilogue: activations + decay blend + 5 output slices --------
#pragma unroll
    for (int mt = 0; mt < 2; ++mt) {
#pragma unroll
        for (int ht = 0; ht < 2; ++ht) {
            const int h   = h0 + hw * 32 + ht * 16 + lane15;
            const float bi  = brec[h];
            const float bz  = brec[2 * Hsz + h];
            const float bo  = brec[3 * Hsz + h];
            const float bib = brec[4 * Hsz + h];
            const float bd  = brec[6 * Hsz + h];
#pragma unroll
            for (int e = 0; e < 8; ++e) {
                const int m  = m0 + mw * 32 + mt * 16 + (khalf << 3) + e;
                const float du = dur[m];
                float xi  = acc[mt][ht][0][e] + bi;
                float xz  = acc[mt][ht][1][e] + bz;
                float xo  = acc[mt][ht][2][e] + bo;
                float xib = acc[mt][ht][3][e] + bib;
                float xd  = acc[mt][ht][4][e] + bd;
                float gi  = fsig(xi);
                float gz  = ftanh(xz);
                float go  = fsig(xo);
                float gib = fsig(xib);
                float gd  = (xd > 20.f) ? xd : __logf(1.f + __expf(xd));
                float c   = gi * gz;
                float cb  = gib * gz;
                float cd  = cb + (c - cb) * __expf(-gd * du);
                float hd  = go * ftanh(cd);
                const int t = m & (Tlen - 1);
                const int b = m >> 10;
                size_t base = (size_t)t * (Bsz * Hsz) + (size_t)b * Hsz + h;
                out[base]                    = hd;
                out[base + (size_t)1 * GOUT] = c;
                out[base + (size_t)2 * GOUT] = cb;
                out[base + (size_t)3 * GOUT] = go;
                out[base + (size_t)4 * GOUT] = gd;
            }
        }
    }
}

extern "C" void kernel_launch(void* const* d_in, const int* in_sizes, int n_in,
                              void* d_out, int out_size, void* d_ws, size_t ws_size,
                              hipStream_t stream) {
    const int*   ev    = (const int*)d_in[0];     // event_seqs   [B,T] int32
    const float* dur   = (const float*)d_in[1];   // duration     [B,T] f32
    const float* W     = (const float*)d_in[2];   // W_rec        [7H,2H] f32
    const float* brec  = (const float*)d_in[3];   // b_rec        [7H] f32
    const float* table = (const float*)d_in[4];   // emb_table    [1025,H] f32
    float*       out   = (float*)d_out;           // [5,T,B,H] f32

    unsigned short* Wb   = (unsigned short*)d_ws;           // NW  bf16
    unsigned short* tabB = Wb + NW;                         // NTAB bf16

    cvt_w<<<(NW + 255) / 256, 256, 0, stream>>>(W, Wb);
    cvt_t<<<(NTAB + 255) / 256, 256, 0, stream>>>(table, tabB, NTAB);

    dim3 grid(MTOT / MT, Hsz / HT);                         // 256 x 4
    size_t lds = (size_t)(2 * BUFSZ) * sizeof(unsigned short); // 221184 B
    ctlstm_fused<<<grid, 512, lds, stream>>>(ev, dur, brec, tabB, Wb, out);
}
